// RNN_15453292331520
// MI455X (gfx1250) — matrix operand order repaired and checked by reading, not compiled, not run on hardware
//
#include <hip/hip_runtime.h>
#include <stdint.h>

typedef __attribute__((ext_vector_type(16))) __bf16 v16bf;
typedef __attribute__((ext_vector_type(8)))  __bf16 v8bf;
typedef __attribute__((ext_vector_type(8)))  float  v8f;

static constexpr int kB = 64;
static constexpr int kS = 512;
static constexpr int kV = 50257;
static constexpr int kE = 512;
static constexpr int kH = 1024;
static constexpr int kK = kE + kH;     // 1536 (both layers' concat width)
static constexpr int NUM_WGS = 32;
static constexpr int SCAN_THREADS = 256;
static constexpr int NT_PROJ = (kV + 15) / 16;   // 3142 column tiles

__device__ __forceinline__ unsigned short f2bf(float f) {
  unsigned u = __float_as_uint(f);
  unsigned r = u + 0x7FFFu + ((u >> 16) & 1u);   // round-to-nearest-even
  return (unsigned short)(r >> 16);
}

__device__ __forceinline__ float fast_tanh(float x) {
#if __has_builtin(__builtin_amdgcn_tanhf)
  return __builtin_amdgcn_tanhf(x);       // gfx1250 V_TANH trans op
#else
  return tanhf(x);
#endif
}

// A-fragment in this code's contiguous k order (B loads match): lane L holds row m=L%16,
// contiguous k [base, base+16) with base = (L/16)*16.
__device__ __forceinline__ v16bf load_a_frag(const unsigned short* row, int kbase) {
  v8bf lo = *(const v8bf*)(row + kbase);
  v8bf hi = *(const v8bf*)(row + kbase + 8);
  return __builtin_shufflevector(lo, hi, 0,1,2,3,4,5,6,7,8,9,10,11,12,13,14,15);
}

__device__ __forceinline__ void grid_barrier(unsigned* bar, unsigned target) {
  __syncthreads();
  if (threadIdx.x == 0) {
    __threadfence();
    atomicAdd(bar, 1u);
    volatile unsigned* vb = (volatile unsigned*)bar;
    while (*vb < target) { __builtin_amdgcn_s_sleep(1); }
    __threadfence();
  }
  __syncthreads();
}

// ---- one-time prep kernels -------------------------------------------------

// dst[n*K + k] = bf16(src[k*N + n])   (transpose + downconvert)
__global__ void k_transpose_bf16(const float* __restrict__ src,
                                 unsigned short* __restrict__ dst, int K, int N) {
  size_t total = (size_t)K * N;
  for (size_t i = (size_t)blockIdx.x * blockDim.x + threadIdx.x; i < total;
       i += (size_t)gridDim.x * blockDim.x) {
    size_t n = i / K, k = i % K;
    dst[i] = f2bf(src[k * (size_t)N + n]);
  }
}

__global__ void k_convert_bf16(const float* __restrict__ src,
                               unsigned short* __restrict__ dst, size_t total) {
  for (size_t i = (size_t)blockIdx.x * blockDim.x + threadIdx.x; i < total;
       i += (size_t)gridDim.x * blockDim.x)
    dst[i] = f2bf(src[i]);
}

__global__ void k_init(const float* __restrict__ ih1, const float* __restrict__ ih2,
                       float* __restrict__ h1f, float* __restrict__ h2f,
                       unsigned short* __restrict__ h1b, unsigned short* __restrict__ h2b,
                       unsigned* __restrict__ bar) {
  size_t i = (size_t)blockIdx.x * blockDim.x + threadIdx.x;
  size_t stride = (size_t)gridDim.x * blockDim.x;
  for (size_t j = i; j < (size_t)kB * kH; j += stride) {
    float v = ih1[j]; h1f[j] = v; h1b[j] = f2bf(v);
  }
  for (size_t j = i; j < (size_t)kB * kE; j += stride) {
    float v = ih2[j]; h2f[j] = v; h2b[j] = f2bf(v);
  }
  if (i == 0) *bar = 0u;
}

// ---- parallel precompute: pre1[t*64+b][n] = emb[seq[b,t]] @ W1[0:512,:] + b1
// Rows r = t*64 + b (32768 rows -> 2048 m-tiles); cols 1024 -> 64 n-tiles.
__global__ void __launch_bounds__(256) pre_kernel(
    const unsigned short* __restrict__ embB,  // [V][512] bf16
    const unsigned short* __restrict__ W1T,   // [1024][1536] bf16 (N-major)
    const int*   __restrict__ seq,            // [64][512]
    const float* __restrict__ b1,
    float* __restrict__ pre1)                 // [S*B][1024] f32
{
  const int wave = threadIdx.x >> 5, lane = threadIdx.x & 31;
  const int tile = blockIdx.x * 8 + wave;     // 131072 tiles total
  const int mtile = tile >> 6;                // consecutive waves share A rows
  const int ntile = tile & 63;
  const int col = lane & 15, kh = lane >> 4;
  const int r0 = mtile * 16, n0 = ntile * 16;

  // per-lane gathered A row: row r = r0+col -> token emb row
  const int r = r0 + col;
  const int t = r >> 6, b = r & 63;
  const int tok = seq[b * kS + t];
  const unsigned short* arow = embB + (size_t)tok * kE;
  const unsigned short* brow = W1T + (size_t)(n0 + col) * kK;   // k 0..511 = token part

  v8f acc = {};
  for (int k0 = 0; k0 < kE; k0 += 32) {
    v16bf a  = load_a_frag(arow, k0 + kh * 16);
    v16bf bm = *(const v16bf*)(brow + k0 + kh * 16);
    acc = __builtin_amdgcn_wmma_f32_16x16x32_bf16(false, a, false, bm,
                                                  (short)0, acc, false, false);
  }
  const float bias = b1[n0 + col];
  for (int rr = 0; rr < 8; ++rr) {
    const int row = r0 + kh * 8 + rr;
    pre1[(size_t)row * kH + n0 + col] = acc[rr] + bias;
  }
}

// ---- persistent recurrent scan --------------------------------------------
// 32 WGs x 256 threads. WG g: m-tile = g%4 (16 batch rows).
// Layer1: N-chunk (g/4)*128, 8 waves x 16 cols, K=1024 (recurrent part only).
// Layer2: N-chunk (g/4)*64, 4 tiles, 2 waves/tile K-split (768 each) + LDS reduce.
__global__ void __launch_bounds__(SCAN_THREADS) rnn_scan(
    const unsigned short* __restrict__ W1T,   // [1024][1536] bf16 (N-major)
    const unsigned short* __restrict__ W2T,   // [512][1536]  bf16 (N-major)
    const float* __restrict__ pre1,           // [S*B][1024] f32 (token part + b1)
    const float* __restrict__ b2,
    float* __restrict__ h1f, float* __restrict__ h2f,
    unsigned short* __restrict__ h1b, unsigned short* __restrict__ h2b,
    unsigned* __restrict__ bar)
{
  __shared__ unsigned short Alds[16 * kK];    // 48 KB A slab (layer1 uses 32 KB)
  __shared__ float red[4 * 32 * 8];           // 4 KB layer2 K-split reduction
  const int tid  = threadIdx.x;
  const int wave = tid >> 5, lane = tid & 31;
  const int g    = blockIdx.x;
  const int mt   = g & 3;
  const int col  = lane & 15;
  const int kh   = lane >> 4;
  unsigned epoch = 0;

  for (int t = 0; t < kS; ++t) {
    const int cur = t & 1, nxt = cur ^ 1;

    // ---- stage A1 = h1_prev rows mt*16..mt*16+15  (16 x 1024 bf16 = 32 KB)
    for (int c = tid; c < 16 * (kH / 8); c += SCAN_THREADS) {
      const int row = c / (kH / 8);
      const int off = (c % (kH / 8)) * 8;
      const int b   = mt * 16 + row;
      *(uint4*)(&Alds[row * kH + off]) =
          *(const uint4*)(h1b + cur * (kB * kH) + b * kH + off);
    }
    __syncthreads();

    // ---- layer 1: new_h1 = tanh(pre1[t] + h1 @ W1[512:,:])
    {
      const int n0 = (g >> 2) * 128 + wave * 16;
      // seed accumulator with precomputed token contribution (+b1)
      v8f acc;
      const float* prow = pre1 + ((size_t)t * kB + mt * 16 + kh * 8) * kH + n0 + col;
      for (int rr = 0; rr < 8; ++rr) acc[rr] = prow[(size_t)rr * kH];

      const unsigned short* arow = &Alds[col * kH];
      const unsigned short* brow = W1T + (size_t)(n0 + col) * kK + kE; // skip token part
      for (int k0 = 0; k0 < kH; k0 += 32) {
        v16bf a  = load_a_frag(arow, k0 + kh * 16);
        v16bf bm = *(const v16bf*)(brow + k0 + kh * 16);
        acc = __builtin_amdgcn_wmma_f32_16x16x32_bf16(false, a, false, bm,
                                                      (short)0, acc, false, false);
      }
      const int mbase = mt * 16 + kh * 8;
      for (int rr = 0; rr < 8; ++rr) {
        const float hv = fast_tanh(acc[rr]);
        const int m = mbase + rr, n = n0 + col;
        h1f[nxt * (kB * kH) + m * kH + n] = hv;
        h1b[nxt * (kB * kH) + m * kH + n] = f2bf(hv);
      }
    }
    grid_barrier(bar, ++epoch * NUM_WGS);

    // ---- stage A2 = [new_h1 | h2_prev]  (16 x 1536 bf16 = 48 KB)
    for (int c = tid; c < 16 * (kK / 8); c += SCAN_THREADS) {
      const int row = c / (kK / 8);
      const int off = (c % (kK / 8)) * 8;
      const int b   = mt * 16 + row;
      const unsigned short* src = (off < kH)
          ? h1b + nxt * (kB * kH) + b * kH + off
          : h2b + cur * (kB * kE) + b * kE + (off - kH);
      *(uint4*)(&Alds[row * kK + off]) = *(const uint4*)src;
    }
    __syncthreads();

    // ---- layer 2: new_h2 = tanh(A2 @ W2 + b2), 2 waves per tile split K
    {
      const int tj   = wave & 3;          // tile within WG
      const int half = wave >> 2;         // K half: 0 -> [0,768), 1 -> [768,1536)
      const int n0   = (g >> 2) * 64 + tj * 16;
      const int kbeg = half * (kK / 2);
      v8f acc = {};
      const unsigned short* arow = &Alds[col * kK];
      const unsigned short* brow = W2T + (size_t)(n0 + col) * kK;
      for (int k0 = kbeg; k0 < kbeg + kK / 2; k0 += 32) {
        v16bf a  = load_a_frag(arow, k0 + kh * 16);
        v16bf bm = *(const v16bf*)(brow + k0 + kh * 16);
        acc = __builtin_amdgcn_wmma_f32_16x16x32_bf16(false, a, false, bm,
                                                      (short)0, acc, false, false);
      }
      if (half) {
        float* s = &red[(tj * 32 + lane) * 8];
        for (int rr = 0; rr < 8; ++rr) s[rr] = acc[rr];
      }
      __syncthreads();
      if (!half) {
        const float* s = &red[(tj * 32 + lane) * 8];
        const float bias = b2[n0 + col];
        const int mbase = mt * 16 + kh * 8;
        for (int rr = 0; rr < 8; ++rr) {
          const float hv = fast_tanh(acc[rr] + s[rr] + bias);
          const int m = mbase + rr, n = n0 + col;
          h2f[nxt * (kB * kE) + m * kE + n] = hv;
          h2b[nxt * (kB * kE) + m * kE + n] = f2bf(hv);
        }
      }
    }
    grid_barrier(bar, ++epoch * NUM_WGS);
  }
}

// ---- tied-embedding output projection: probs = h2 @ emb^T + out_b ---------
__global__ void __launch_bounds__(256) proj_kernel(
    const unsigned short* __restrict__ h2bf,  // [64][512] bf16 (final, buffer 0)
    const unsigned short* __restrict__ embB,  // [50257][512] bf16
    const float* __restrict__ out_b,
    float* __restrict__ out)                  // [64][50257] f32
{
  const int wave = threadIdx.x >> 5, lane = threadIdx.x & 31;
  const int tile = blockIdx.x * 8 + wave;
  if (tile >= 4 * NT_PROJ) return;            // wave-uniform
  const int mtile = tile & 3;
  const int ntile = tile >> 2;
  const int col = lane & 15, kh = lane >> 4;
  const int v = ntile * 16 + col;
  const bool valid = (v < kV);
  const int vc = valid ? v : 0;

  v8f acc = {};
  const unsigned short* arow = h2bf + (size_t)(mtile * 16 + col) * kE;
  const unsigned short* brow = embB + (size_t)vc * kE;  // B[k,n]=emb[v,k]: row-contig
  for (int k0 = 0; k0 < kE; k0 += 32) {
    v16bf a  = load_a_frag(arow, k0 + kh * 16);
    v16bf bm = *(const v16bf*)(brow + k0 + kh * 16);
    acc = __builtin_amdgcn_wmma_f32_16x16x32_bf16(false, a, false, bm,
                                                  (short)0, acc, false, false);
  }
  if (valid) {
    const float bias = out_b[v];
    for (int rr = 0; rr < 8; ++rr) {
      const int m = mtile * 16 + kh * 8 + rr;
      out[(size_t)m * kV + v] = acc[rr] + bias;
    }
  }
}

__global__ void k_tail(const float* __restrict__ h1f0, const float* __restrict__ h2f0,
                       float* __restrict__ out) {
  const size_t base1 = (size_t)kB * kV;
  const size_t base2 = base1 + (size_t)kB * kH;
  size_t i = (size_t)blockIdx.x * blockDim.x + threadIdx.x;
  size_t stride = (size_t)gridDim.x * blockDim.x;
  for (size_t j = i; j < (size_t)kB * kH; j += stride) out[base1 + j] = h1f0[j];
  for (size_t j = i; j < (size_t)kB * kE; j += stride) out[base2 + j] = h2f0[j];
}

// ---- host side -------------------------------------------------------------
extern "C" void kernel_launch(void* const* d_in, const int* in_sizes, int n_in,
                              void* d_out, int out_size, void* d_ws, size_t ws_size,
                              hipStream_t stream) {
  (void)in_sizes; (void)n_in; (void)out_size; (void)ws_size;
  const int*   seq     = (const int*)  d_in[0];
  const float* init_h1 = (const float*)d_in[1];
  const float* init_h2 = (const float*)d_in[2];
  const float* emb_W   = (const float*)d_in[3];
  const float* W1      = (const float*)d_in[4];
  const float* b1      = (const float*)d_in[5];
  const float* W2      = (const float*)d_in[6];
  const float* b2      = (const float*)d_in[7];
  const float* out_b   = (const float*)d_in[8];
  float* out = (float*)d_out;

  char* ws = (char*)d_ws;
  size_t off = 0;
  auto alloc = [&](size_t bytes) -> char* {
    char* p = ws + off;
    off += (bytes + 255) & ~(size_t)255;
    return p;
  };
  unsigned short* W1T  = (unsigned short*)alloc((size_t)kH * kK * 2);  // 3.0 MB
  unsigned short* W2T  = (unsigned short*)alloc((size_t)kE * kK * 2);  // 1.5 MB
  unsigned short* embB = (unsigned short*)alloc((size_t)kV * kE * 2);  // 49 MB
  float* h1f = (float*)alloc((size_t)2 * kB * kH * 4);
  float* h2f = (float*)alloc((size_t)2 * kB * kE * 4);
  unsigned short* h1b = (unsigned short*)alloc((size_t)2 * kB * kH * 2);
  unsigned short* h2b = (unsigned short*)alloc((size_t)2 * kB * kE * 2);
  unsigned* bar = (unsigned*)alloc(256);
  float* pre1 = (float*)alloc((size_t)kS * kB * kH * 4);               // 128 MB

  k_transpose_bf16<<<1024, 256, 0, stream>>>(W1, W1T, kK, kH);
  k_transpose_bf16<<<512,  256, 0, stream>>>(W2, W2T, kK, kE);
  k_convert_bf16<<<2048, 256, 0, stream>>>(emb_W, embB, (size_t)kV * kE);
  k_init<<<256, 256, 0, stream>>>(init_h1, init_h2, h1f, h2f, h1b, h2b, bar);

  // token contribution for all 512 steps: 131072 tiles, 8 waves/block
  pre_kernel<<<131072 / 8, 256, 0, stream>>>(embB, W1T, seq, b1, pre1);

  rnn_scan<<<NUM_WGS, SCAN_THREADS, 0, stream>>>(W1T, W2T, pre1, b2,
                                                 h1f, h2f, h1b, h2b, bar);

  // 4*3142 = 12568 tiles, 8 waves/block -> exactly 1571 blocks
  proj_kernel<<<(4 * NT_PROJ) / 8, 256, 0, stream>>>(h2b, embB, out_b, out);
  k_tail<<<64, 256, 0, stream>>>(h1f, h2f, out);  // final states live in buffer 0
}
